// MultiHeadAttention_27797028339762
// MI455X (gfx1250) — matrix operand order repaired and checked by reading, not compiled, not run on hardware
//
#include <hip/hip_runtime.h>
#include <hip/hip_bf16.h>
#include <math.h>

// ---------------------------------------------------------------------------
// MHA forward for B=1, S=4096, H=1024, NH=16, HD=64 on gfx1250 (MI455X).
// All matmuls use v_wmma_f32_16x16x32_f16 (f16 operands, f32 accumulate).
// GEMM A-tiles staged with GLOBAL_LOAD_ASYNC_TO_LDS_B128 (ASYNCcnt path).
// ---------------------------------------------------------------------------

constexpr int SEQ   = 4096;
constexpr int HID   = 1024;
constexpr int NHEAD = 16;
constexpr int HDIM  = 64;

typedef _Float16 v4h  __attribute__((ext_vector_type(4)));
typedef _Float16 v8h  __attribute__((ext_vector_type(8)));
typedef _Float16 v16h __attribute__((ext_vector_type(16)));
typedef float    v8f  __attribute__((ext_vector_type(8)));

__device__ __forceinline__ v16h cat8(v8h lo, v8h hi) {
  return __builtin_shufflevector(lo, hi, 0,1,2,3,4,5,6,7,8,9,10,11,12,13,14,15);
}

// A-fragment (16x32 f16) register layout (contiguous K order, matches B):
// lane<16 (row = lane): halves 0..15 = K 0..15
// lane>=16 (row = lane-16): halves 0..15 = K 16..31
__device__ __forceinline__ v16h load_afrag(const _Float16* rowp, int kh) {
  v8h lo = *(const v8h*)(rowp + kh * 16);
  v8h hi = *(const v8h*)(rowp + kh * 16 + 8);
  return cat8(lo, hi);
}

__device__ __forceinline__ v8f wmma_f16(v16h a, v16h b, v8f c) {
  return __builtin_amdgcn_wmma_f32_16x16x32_f16(false, a, false, b, (short)0, c,
                                                false, false);
}

// Async copy of 64 contiguous bytes global -> LDS (4 x b128, shared offset
// applies to both global source and LDS destination per CDNA5 ISA 15.18.3).
__device__ __forceinline__ void async_copy64(const void* gsrc, void* ldst) {
  unsigned laddr = (unsigned)(uintptr_t)ldst;   // low 32 bits = LDS byte addr
  asm volatile(
      "global_load_async_to_lds_b128 %0, %1, off\n\t"
      "global_load_async_to_lds_b128 %0, %1, off offset:16\n\t"
      "global_load_async_to_lds_b128 %0, %1, off offset:32\n\t"
      "global_load_async_to_lds_b128 %0, %1, off offset:48"
      :: "v"(laddr), "v"(gsrc) : "memory");
}

__device__ __forceinline__ void wait_async0() {
  asm volatile("s_wait_asynccnt 0" ::: "memory");
}

// ---------------------------------------------------------------------------
// fp32 -> fp16 convert (vectorized: float4 -> 4 halves)
// ---------------------------------------------------------------------------
__global__ __launch_bounds__(256) void cvt_f32_to_f16(const float* __restrict__ x,
                                                      _Float16* __restrict__ y,
                                                      int n4) {
  int i = blockIdx.x * 256 + threadIdx.x;
  if (i < n4) {
    float4 f = ((const float4*)x)[i];
    v4h h = {(_Float16)f.x, (_Float16)f.y, (_Float16)f.z, (_Float16)f.w};
    ((v4h*)y)[i] = h;
  }
}

// ---------------------------------------------------------------------------
// C[M,N] = A[M,K](f16) @ W[N,K]^T (W f32, converted while staging to LDS).
// Block = 128x128, 256 threads = 8 waves (2x4), wave tile 64x32 (4x2 WMMA).
// TK = 64 with LDS double buffering; A staged via async-to-LDS.
// ---------------------------------------------------------------------------
template <bool OUTF32>
__global__ __launch_bounds__(256) void gemm_xwT(const _Float16* __restrict__ A,
                                                const float* __restrict__ W,
                                                void* __restrict__ outp) {
  constexpr int N = HID, K = HID;

  __shared__ alignas(128) _Float16 As[2][128 * 64];
  __shared__ alignas(128) _Float16 Ws[2][128 * 64];

  const int t    = threadIdx.x;
  const int lane = t & 31;
  const int wave = t >> 5;
  const int wm   = wave >> 2;       // 0..1  (row group of 64)
  const int wn   = wave & 3;        // 0..3  (col group of 32)
  const int kh   = lane >> 4;
  const int nrow = lane & 15;

  const int rowBase = blockIdx.y * 128;
  const int colBase = blockIdx.x * 128;

  const int srow = t >> 1;          // 0..127 staged row
  const int skO  = (t & 1) * 32;    // 0 or 32 within 64-wide K panel

  const _Float16* aball = &A[(size_t)(rowBase + srow) * K + skO];
  const float*    wball = &W[(size_t)(colBase + srow) * K + skO];

  v8f acc[4][2] = {};

  auto stage = [&](int buf, int kk) {
    // A tile: 64B per thread, async global -> LDS (no VGPR round-trip)
    async_copy64(aball + kk, &As[buf][srow * 64 + skO]);
    // W tile: f32 -> f16 conversion must go through the VALU
    if (kk + 128 < K) __builtin_prefetch(wball + kk + 128, 0, 3);
    const float* wsrc = wball + kk;
    _Float16* wdst = &Ws[buf][srow * 64 + skO];
#pragma unroll
    for (int i = 0; i < 32; i += 4) {
      float4 f = *(const float4*)(wsrc + i);
      v4h h = {(_Float16)f.x, (_Float16)f.y, (_Float16)f.z, (_Float16)f.w};
      *(v4h*)(wdst + i) = h;
    }
  };

  auto compute = [&](int buf) {
#pragma unroll
    for (int ks = 0; ks < 2; ++ks) {
      v16h af[4];
#pragma unroll
      for (int mt = 0; mt < 4; ++mt)
        af[mt] = load_afrag(&As[buf][(wm * 64 + mt * 16 + nrow) * 64 + ks * 32], kh);
      v16h bf[2];
#pragma unroll
      for (int nt = 0; nt < 2; ++nt)
        bf[nt] = *(const v16h*)&Ws[buf][(wn * 32 + nt * 16 + nrow) * 64 + ks * 32 + kh * 16];
#pragma unroll
      for (int mt = 0; mt < 4; ++mt)
#pragma unroll
        for (int nt = 0; nt < 2; ++nt)
          acc[mt][nt] = wmma_f16(af[mt], bf[nt], acc[mt][nt]);
    }
  };

  stage(0, 0);
  wait_async0();
  __syncthreads();
  for (int kk = 0; kk < K; kk += 128) {
    stage(1, kk + 64);        // async transfers fly under compute(0)
    compute(0);
    wait_async0();
    __syncthreads();
    if (kk + 128 < K) stage(0, kk + 128);
    compute(1);
    wait_async0();
    __syncthreads();
  }

  // C layout: VGPR v -> row v (lanes 0-15) / v+8 (lanes 16-31), col = lane&15
  const int hi8 = kh << 3;
#pragma unroll
  for (int mt = 0; mt < 4; ++mt)
#pragma unroll
    for (int nt = 0; nt < 2; ++nt) {
      const size_t base = (size_t)(rowBase + wm * 64 + mt * 16 + hi8) * N +
                          (colBase + wn * 32 + nt * 16 + nrow);
      if constexpr (OUTF32) {
        float* out = (float*)outp;
#pragma unroll
        for (int v = 0; v < 8; ++v)
          out[base + (size_t)v * N] = acc[mt][nt][v];
      } else {
        _Float16* out = (_Float16*)outp;
#pragma unroll
        for (int v = 0; v < 8; ++v)
          out[base + (size_t)v * N] = (_Float16)acc[mt][nt][v];
      }
    }
}

// ---------------------------------------------------------------------------
// RoPE on Q/K + layout change: [S][H] -> Q,K [NH][S][HD], V -> [NH][HD][S]
// ---------------------------------------------------------------------------
__global__ __launch_bounds__(256) void rope_relayout(const _Float16* __restrict__ Qh,
                                                     const _Float16* __restrict__ Kh,
                                                     const _Float16* __restrict__ Vh,
                                                     _Float16* __restrict__ Qr,
                                                     _Float16* __restrict__ Kr,
                                                     _Float16* __restrict__ Vt) {
  const int s = blockIdx.x;
  for (int p = threadIdx.x; p < NHEAD * 32; p += 256) {
    const int h = p >> 5;        // head
    const int j = p & 31;        // rotation pair index, 0..31
    const float inv  = expf(-(float)(2 * j) / 64.0f * 9.210340372f); // ln(1e4)
    const float ang  = (float)s * inv;
    const float c    = cosf(ang);
    const float sn   = sinf(ang);

    const size_t src1 = (size_t)s * HID + h * HDIM + j;
    const size_t src2 = src1 + 32;
    const size_t dst  = ((size_t)h * SEQ + s) * HDIM + j;

    float q1 = (float)Qh[src1], q2 = (float)Qh[src2];
    Qr[dst]      = (_Float16)(q1 * c - q2 * sn);
    Qr[dst + 32] = (_Float16)(q2 * c + q1 * sn);

    float k1 = (float)Kh[src1], k2 = (float)Kh[src2];
    Kr[dst]      = (_Float16)(k1 * c - k2 * sn);
    Kr[dst + 32] = (_Float16)(k2 * c + k1 * sn);

    Vt[((size_t)(h * HDIM + j)) * SEQ + s]      = Vh[src1];
    Vt[((size_t)(h * HDIM + j + 32)) * SEQ + s] = Vh[src2];
  }
}

// ---------------------------------------------------------------------------
// Flash attention: grid (SEQ/64, NHEAD), 128 threads = 4 waves x 16 q-rows.
// Online softmax; P re-shaped C-layout -> A-fragment via per-wave LDS tile.
// ---------------------------------------------------------------------------
__global__ __launch_bounds__(128) void attn_flash(const _Float16* __restrict__ Qr,
                                                  const _Float16* __restrict__ Kr,
                                                  const _Float16* __restrict__ Vt,
                                                  const int* __restrict__ amask,
                                                  _Float16* __restrict__ AO) {
  __shared__ alignas(64) _Float16 Plds[4][16][72];   // 72-halve row stride

  const int lane = threadIdx.x & 31;
  const int w    = threadIdx.x >> 5;
  const int kh   = lane >> 4;
  const int nrow = lane & 15;
  const int hi8  = kh << 3;
  const int head = blockIdx.y;
  const int qbase = blockIdx.x * 64 + w * 16;

  // Q fragments for this wave's 16 rows (two K-steps covering HD=64)
  const int qrow = qbase + nrow;
  v16h aq[2];
#pragma unroll
  for (int ks = 0; ks < 2; ++ks)
    aq[ks] = load_afrag(Qr + ((size_t)head * SEQ + qrow) * HDIM + ks * 32, kh);

  float mrow[8], lrow[8];
  v8f o[4] = {};
#pragma unroll
  for (int v = 0; v < 8; ++v) { mrow[v] = -1e30f; lrow[v] = 0.0f; }

  const float scale = 0.125f;  // 1/sqrt(64)
  const _Float16* krowp = Kr + ((size_t)head * SEQ + nrow) * HDIM + kh * 16;

  for (int kb = 0; kb < SEQ; kb += 64) {
    // prefetch next key block rows
    if (kb + 64 < SEQ) __builtin_prefetch(krowp + (size_t)(kb + 64) * HDIM, 0, 3);

    // ---- scores: 16 q-rows x 64 keys, 4 WMMA tiles, 2 K-steps of 32 ----
    v8f st[4] = {};
#pragma unroll
    for (int kt = 0; kt < 4; ++kt) {
      const _Float16* kp = krowp + (size_t)(kb + kt * 16) * HDIM;
#pragma unroll
      for (int ks = 0; ks < 2; ++ks) {
        v16h bk = *(const v16h*)(kp + ks * 32);
        st[kt] = wmma_f16(aq[ks], bk, st[kt]);
      }
      const bool ok = amask[kb + kt * 16 + nrow] != 0;
#pragma unroll
      for (int v = 0; v < 8; ++v)
        st[kt][v] = ok ? st[kt][v] * scale : -1e30f;
    }

    // ---- online softmax (row v lives in lanes 0-15, row v+8 in 16-31) ----
#pragma unroll
    for (int v = 0; v < 8; ++v) {
      float x = fmaxf(fmaxf(st[0][v], st[1][v]), fmaxf(st[2][v], st[3][v]));
      x = fmaxf(x, __shfl_xor(x, 1));
      x = fmaxf(x, __shfl_xor(x, 2));
      x = fmaxf(x, __shfl_xor(x, 4));
      x = fmaxf(x, __shfl_xor(x, 8));
      const float nm = fmaxf(mrow[v], x);
      const float alpha = __expf(mrow[v] - nm);
      float rs = 0.0f;
#pragma unroll
      for (int kt = 0; kt < 4; ++kt) {
        float e = __expf(st[kt][v] - nm);
        st[kt][v] = e;
        rs += e;
      }
      rs += __shfl_xor(rs, 1);
      rs += __shfl_xor(rs, 2);
      rs += __shfl_xor(rs, 4);
      rs += __shfl_xor(rs, 8);
      lrow[v] = lrow[v] * alpha + rs;
      mrow[v] = nm;
#pragma unroll
      for (int nt = 0; nt < 4; ++nt) o[nt][v] *= alpha;
    }

    // ---- P (C-layout) -> LDS -> A-fragment layout ----
#pragma unroll
    for (int kt = 0; kt < 4; ++kt)
#pragma unroll
      for (int v = 0; v < 8; ++v)
        Plds[w][v + hi8][kt * 16 + nrow] = (_Float16)st[kt][v];
    // intra-wave LDS RAW: ensure stores land before fragment reloads
    asm volatile("s_wait_dscnt 0" ::: "memory");

    // ---- O += P @ V  (keys contracted, 2 K-steps; 4 hd tiles of 16) ----
#pragma unroll
    for (int ks = 0; ks < 2; ++ks) {
      v16h ap = load_afrag(&Plds[w][nrow][ks * 32], kh);
#pragma unroll
      for (int nt = 0; nt < 4; ++nt) {
        const _Float16* vp = Vt + ((size_t)(head * HDIM + nt * 16 + nrow)) * SEQ
                              + kb + ks * 32 + kh * 16;
        v16h bv = *(const v16h*)vp;
        o[nt] = wmma_f16(ap, bv, o[nt]);
      }
    }
    asm volatile("" ::: "memory");   // WAR: reads done before next P store
  }

  // ---- finalize: divide by row sum, write f16 [S][H] ----
#pragma unroll
  for (int nt = 0; nt < 4; ++nt) {
    const size_t base = (size_t)(qbase + hi8) * HID + head * HDIM + nt * 16 + nrow;
#pragma unroll
    for (int v = 0; v < 8; ++v)
      AO[base + (size_t)v * HID] = (_Float16)(o[nt][v] / lrow[v]);
  }
}

// ---------------------------------------------------------------------------
// Launcher
// ---------------------------------------------------------------------------
extern "C" void kernel_launch(void* const* d_in, const int* in_sizes, int n_in,
                              void* d_out, int out_size, void* d_ws, size_t ws_size,
                              hipStream_t stream) {
  const float* X    = (const float*)d_in[0];
  const int*   mask = (const int*)d_in[1];
  const float* Wq   = (const float*)d_in[2];
  const float* Wk   = (const float*)d_in[3];
  const float* Wv   = (const float*)d_in[4];
  const float* Wo   = (const float*)d_in[5];
  float* out = (float*)d_out;

  // workspace layout (f16 buffers, 8 MiB each = SEQ*HID*2 bytes)
  char* ws = (char*)d_ws;
  const size_t SZ = (size_t)SEQ * HID * sizeof(_Float16);
  _Float16* Xh = (_Float16*)(ws + 0 * SZ);
  _Float16* Qh = (_Float16*)(ws + 1 * SZ);
  _Float16* Kh = (_Float16*)(ws + 2 * SZ);
  _Float16* Vh = (_Float16*)(ws + 3 * SZ);
  _Float16* Qr = (_Float16*)(ws + 4 * SZ);
  _Float16* Kr = (_Float16*)(ws + 5 * SZ);
  _Float16* Vt = (_Float16*)(ws + 6 * SZ);
  _Float16* AO = (_Float16*)(ws + 1 * SZ);  // reuse Qh (dead after rope)

  const int n4 = SEQ * HID / 4;
  cvt_f32_to_f16<<<(n4 + 255) / 256, 256, 0, stream>>>(X, Xh, n4);

  dim3 gg(HID / 128, SEQ / 128);  // (8, 32)
  gemm_xwT<false><<<gg, 256, 0, stream>>>(Xh, Wq, Qh);
  gemm_xwT<false><<<gg, 256, 0, stream>>>(Xh, Wk, Kh);
  gemm_xwT<false><<<gg, 256, 0, stream>>>(Xh, Wv, Vh);

  rope_relayout<<<SEQ, 256, 0, stream>>>(Qh, Kh, Vh, Qr, Kr, Vt);

  attn_flash<<<dim3(SEQ / 64, NHEAD), 128, 0, stream>>>(Qr, Kr, Vt, mask, AO);

  gemm_xwT<true><<<gg, 256, 0, stream>>>(AO, Wo, out);
}
